// Graph_Learn_24739011625001
// MI455X (gfx1250) — compile-verified, matrix operand order checked by reading, not run on hardware
//
#include <hip/hip_runtime.h>

typedef __attribute__((ext_vector_type(16))) _Float16 v16h;
typedef __attribute__((ext_vector_type(8)))  float    v8f;

#define NB 8      // batch N
#define TT 8      // time T
#define VV 512    // nodes V
#define FF 64     // features F

// Pass 1: tmpS[n,i,j] = exp(relu(sum_f a[f]*|xm[n,i,f]-xm[n,j,f]|)) -> d_out
//         colsum[n,j] = sum_i tmpS[n,i,j]                            -> d_ws
// One block per (j-tile of 16, n). 8 waves; each wave owns two 32-row i-blocks.
// Each lane owns one column j (lanes l and l+16 share j = tile*16 + (l&15) and
// split the 32 i's per block per the WMMA 16-bit A-matrix layout:
//   lanes 0-15 : K = 0..7, 16..23 ; lanes 16-31 : K = 8..15, 24..31
// so the f16 tmpS tile sits directly in A (M=j, K=i) and
// v_wmma_f32_16x16x32_f16 with an all-ones B accumulates the column sums
// into an f32 C across i-blocks (replicated over N; extracted from lanes 0/16).
__global__ __launch_bounds__(256)
void gl_tmps_colsum(const float* __restrict__ x,
                    const float* __restrict__ a,
                    float* __restrict__ out,
                    float* __restrict__ colsum) {
  const int lane = threadIdx.x & 31;
  const int wave = threadIdx.x >> 5;
  const int jt   = blockIdx.x;        // 0..31
  const int n    = blockIdx.y;        // 0..7
  const int jl   = lane & 15;
  const int half = lane >> 4;
  const int jcol = jt * 16 + jl;

  const float* xm = x + (((size_t)n * TT + TT / 2) * VV) * FF;  // x[n, T/2, :, :]

  // Per-lane: its own column row of xm, plus the weight vector, in registers.
  float4 xj[16], av[16];
  const float4* xjp = (const float4*)(xm + (size_t)jcol * FF);
  const float4* ap  = (const float4*)a;
#pragma unroll
  for (int q = 0; q < 16; ++q) { xj[q] = xjp[q]; av[q] = ap[q]; }

  v8f cacc = {};          // WMMA C: column-sum accumulator (f32)
  v16h ones;
#pragma unroll
  for (int e = 0; e < 16; ++e) ones[e] = (_Float16)1.0f;  // layout-proof B

  for (int t = 0; t < 2; ++t) {
    const int ib = (t * 8 + wave) * 32;   // this wave's 32-row i-block
    __builtin_prefetch(xm + (size_t)(ib + 32) * FF, 0, 0);  // global_prefetch_b8
    v16h afrag;
#pragma unroll
    for (int k = 0; k < 16; ++k) {
      const int g = k >> 3, m = k & 7;
      const int i = ib + half * 8 + g * 16 + m;   // matches A-matrix K ordering
      const float4* xip = (const float4*)(xm + (size_t)i * FF);
      float acc = 0.f;
#pragma unroll
      for (int q = 0; q < 16; ++q) {              // uniform addr -> broadcast load
        float4 xv = xip[q];
        acc = fmaf(fabsf(xv.x - xj[q].x), av[q].x, acc);
        acc = fmaf(fabsf(xv.y - xj[q].y), av[q].y, acc);
        acc = fmaf(fabsf(xv.z - xj[q].z), av[q].z, acc);
        acc = fmaf(fabsf(xv.w - xj[q].w), av[q].w, acc);
      }
      float tmp = __expf(fmaxf(acc, 0.f));
      out[((size_t)n * VV + i) * VV + jcol] = tmp;   // exact f32 tmpS
      afrag[k] = (_Float16)tmp;                      // f16 copy only for colsum
    }
    // D[m,n] (+=) sum_K A[m,K] : per-column sums over this 32-row i-block.
    cacc = __builtin_amdgcn_wmma_f32_16x16x32_f16(false, afrag, false, ones,
                                                  (short)0, cacc, false, false);
  }

  // C/D layout: VGPR r holds M=r on lanes 0-15 and M=r+8 on lanes 16-31,
  // replicated over N -> lane 0 carries colsum[j=jt*16+r], lane 16 carries +8.
  __shared__ float wsum[8][16];
  if (jl == 0) {
#pragma unroll
    for (int r = 0; r < 8; ++r) wsum[wave][half * 8 + r] = cacc[r];
  }
  __syncthreads();
  if (threadIdx.x < 16) {         // deterministic fixed-order block reduction
    float s = 0.f;
#pragma unroll
    for (int w = 0; w < 8; ++w) s += wsum[w][threadIdx.x];
    colsum[((size_t)n << 9) + jt * 16 + threadIdx.x] = s;
  }
}

// Pass 2: S[n,i,j] = tmpS[n,i,j] / colsum[n,j]
__global__ __launch_bounds__(256)
void gl_normalize(float* __restrict__ out, const float* __restrict__ colsum) {
  const int idx = blockIdx.x * 256 + threadIdx.x;   // 8*512*512 total
  const int n = idx >> 18;                          // 512*512 = 2^18
  const int j = idx & (VV - 1);
  out[idx] = out[idx] / colsum[(n << 9) + j];
}

extern "C" void kernel_launch(void* const* d_in, const int* in_sizes, int n_in,
                              void* d_out, int out_size, void* d_ws, size_t ws_size,
                              hipStream_t stream) {
  const float* x = (const float*)d_in[0];   // (8,8,512,64) f32
  const float* a = (const float*)d_in[1];   // (64,1) f32
  float* out    = (float*)d_out;            // (8,512,512) f32
  float* colsum = (float*)d_ws;             // 8*512 f32 scratch (fully overwritten)

  dim3 g1(VV / 16, NB);
  gl_tmps_colsum<<<g1, 256, 0, stream>>>(x, a, out, colsum);

  const int total = NB * VV * VV;
  gl_normalize<<<total / 256, 256, 0, stream>>>(out, colsum);
}
